// MultiScaleWalkSampler_84988812853329
// MI455X (gfx1250) — compile-verified, matrix operand order checked by reading, not run on hardware
//
#include <hip/hip_runtime.h>
#include <hip/hip_bf16.h>
#include <cstdint>
#include <math.h>

typedef __attribute__((ext_vector_type(16))) _Float16 v16h;
typedef __attribute__((ext_vector_type(8)))  float    v8f;

#define NUM_NODES 200000
#define MAX_DEG   32
#define BATCH     8192
#define MEM_DIM   128
#define TIME_DIM  64
#define INV_TAU   10.0f
#define NEGF      (-1e30f)

#define W_S 10
#define L_S 3
#define W_L 5
#define L_L 10
#define W_T 5
#define L_T 8

// ---------------- deterministic RNG (counter-hash) ----------------
__device__ __forceinline__ uint32_t pcg(uint32_t x) {
  x = x * 747796405u + 2891336453u;
  uint32_t w = ((x >> ((x >> 28u) + 4u)) ^ x) * 277803737u;
  return (w >> 22u) ^ w;
}
__device__ __forceinline__ float u01(uint32_t h) {
  return ((float)(h >> 8) + 0.5f) * (1.0f / 16777216.0f);
}
__device__ __forceinline__ float gumbel(uint32_t h) {
  float u = u01(h);
  return -__logf(-__logf(u));
}

// A/B fragment K mapping for 16-bit WMMA operands (ISA 7.12.2):
//   low lane (variant 0): comp i -> K = i<8 ? i   : i+8
//   high lane (variant 1): comp i -> K = i<8 ? i+8 : i+16
__device__ __forceinline__ int kmap(int i, int v) {
  return (i < 8) ? (i + (v ? 8 : 0)) : (i + (v ? 16 : 8));
}

// =================================================================
// Plain temporal-biased walks: one wave32 per walk, one lane per
// neighbor (MAX_DEG == 32 == wave width). Gumbel-max categorical.
// =================================================================
__global__ void __launch_bounds__(256)
plain_walks_kernel(const int* __restrict__ src,
                   const float* __restrict__ t0,
                   const int* __restrict__ nb_ids,
                   const float* __restrict__ nb_ts,
                   float* __restrict__ out_nodes,
                   float* __restrict__ out_times,
                   float* __restrict__ out_masks,
                   int W, int L, int total_walks, uint32_t seed) {
  const int tid  = blockIdx.x * blockDim.x + threadIdx.x;
  const int wi   = tid >> 5;     // wave index == walk index
  const int lane = tid & 31;
  if (wi >= total_walks) return;

  const int   b   = wi / W;
  const int   n0  = src[b];
  const float t0v = t0[b];

  int   cur_n = n0;
  float cur_t = t0v;
  bool  alive = true;

  if (lane == 0) {
    size_t o = (size_t)wi * L;
    out_nodes[o] = (float)n0;
    out_times[o] = t0v;
    out_masks[o] = 1.0f;
  }

  for (int step = 1; step < L; ++step) {
    const size_t row = (size_t)cur_n * MAX_DEG + lane;
    float nt = nb_ts[row];
    int   ni = nb_ids[row];
    bool  valid = nt < cur_t;

    uint32_t h = pcg(pcg(pcg(seed + (uint32_t)wi) + (uint32_t)step) + (uint32_t)lane);
    float key = valid ? fmaf(nt, INV_TAU, gumbel(h)) : NEGF;

    // wave argmax carrying (key, id, time, lane-index)
    float bk = key; int bid = ni; float btm = nt; int bidx = lane;
    #pragma unroll
    for (int off = 16; off > 0; off >>= 1) {
      float ok   = __shfl_xor(bk,   off, 32);
      int   oid  = __shfl_xor(bid,  off, 32);
      float otm  = __shfl_xor(btm,  off, 32);
      int   oidx = __shfl_xor(bidx, off, 32);
      if (ok > bk || (ok == bk && oidx < bidx)) { bk = ok; bid = oid; btm = otm; bidx = oidx; }
    }

    bool has = (__ballot(valid) != 0ull);
    alive = alive && has;
    if (alive) { cur_n = bid; cur_t = btm; }

    if (lane == 0) {
      size_t o = (size_t)wi * L + step;
      out_nodes[o] = alive ? (float)cur_n : 0.0f;
      out_times[o] = alive ? cur_t        : 0.0f;
      out_masks[o] = alive ? 1.0f         : 0.0f;
    }
  }
}

// =================================================================
// TAWR walks: 16 walks per wave. The 192-dim restart GEMV
// rho = sigmoid(feat . w + b) is computed with v_wmma_f32_16x16x32_f16
// (A = 16 walks x 32-K feature chunk, B = restart_w chunk replicated
// across all 16 columns, 6 chunks accumulated).
//
// __launch_bounds__(256, 1): lift the occupancy-driven VGPR cap so the
// 48 VGPRs of hoisted B fragments stay resident (no scratch spills in
// the hot loop). The TAWR launch is only 2560 waves chip-wide, so the
// nominal occupancy loss is irrelevant; scratch round-trips are not.
// Grid is exactly 2560 full waves -> EXEC is all-ones at every WMMA.
// =================================================================
__global__ void __launch_bounds__(256, 1)
tawr_walks_kernel(const int* __restrict__ src,
                  const float* __restrict__ t0arr,
                  const int* __restrict__ nb_ids,
                  const float* __restrict__ nb_ts,
                  const float* __restrict__ mem,
                  const float* __restrict__ rw,
                  const float* __restrict__ rb_,
                  const float* __restrict__ tw,
                  const float* __restrict__ tb,
                  float* __restrict__ out_nodes,
                  float* __restrict__ out_times,
                  float* __restrict__ out_masks,
                  float* __restrict__ out_rflag,
                  float* __restrict__ out_rho,
                  uint32_t seed) {
  const int tid   = blockIdx.x * blockDim.x + threadIdx.x;
  const int lt    = threadIdx.x;
  const int wave  = tid >> 5;
  const int lane  = tid & 31;
  const int w     = lane & 15;              // walk-in-wave (0..15)
  const bool hi   = lane >= 16;             // high half-lane of the pair
  const int g     = wave * 16 + w;          // global TAWR walk id
  const int b     = g / W_T;

  // ---- stage the 2 B-fragment variants (6 chunks x 16 halfs) in LDS ----
  __shared__ __align__(32) _Float16 bwLDS[6][2][16];
  if (lt < 192) {
    const int c = lt >> 5;          // chunk 0..5
    const int v = (lt >> 4) & 1;    // variant 0(low)/1(high)
    const int i = lt & 15;          // component 0..15
    bwLDS[c][v][i] = (_Float16)rw[32 * c + kmap(i, v)];
  }
  __syncthreads();
  const v16h* bwrow = (const v16h*)&bwLDS[0][hi ? 1 : 0][0];  // stride 2 v16h per chunk

  const int   n0  = src[b];
  const float t0v = t0arr[b];
  const float restart_b = rb_[0];

  int   cur_n = n0;
  float cur_t = t0v;
  bool  alive = true;

  if (!hi) {
    size_t o = (size_t)g * L_T;
    out_nodes[o] = (float)n0;
    out_times[o] = t0v;
    out_masks[o] = 1.0f;
    out_rflag[o] = 0.0f;
    out_rho[o]   = 0.0f;
  }

  for (int step = 1; step < L_T; ++step) {
    // ---- restart rho via WMMA over feat = [mem(128) || cos(t*tw+tb)(64)] ----
    const float* memrow = mem + (size_t)cur_n * MEM_DIM;
    v8f acc = {};
    // chunks 0..3: memory part (contiguous 8-float runs -> b128 loads)
    #pragma unroll
    for (int c = 0; c < 4; ++c) {
      const int rA = 32 * c + (hi ? 8 : 0);
      const int rB = rA + 16;
      float4 a0 = *(const float4*)(memrow + rA);
      float4 a1 = *(const float4*)(memrow + rA + 4);
      float4 b0 = *(const float4*)(memrow + rB);
      float4 b1 = *(const float4*)(memrow + rB + 4);
      v16h a;
      a[0]=(_Float16)a0.x; a[1]=(_Float16)a0.y; a[2]=(_Float16)a0.z; a[3]=(_Float16)a0.w;
      a[4]=(_Float16)a1.x; a[5]=(_Float16)a1.y; a[6]=(_Float16)a1.z; a[7]=(_Float16)a1.w;
      a[8]=(_Float16)b0.x; a[9]=(_Float16)b0.y; a[10]=(_Float16)b0.z; a[11]=(_Float16)b0.w;
      a[12]=(_Float16)b1.x; a[13]=(_Float16)b1.y; a[14]=(_Float16)b1.z; a[15]=(_Float16)b1.w;
      v16h bc = bwrow[2 * c];   // constant B fragment (regs after hoist, LDS else)
      acc = __builtin_amdgcn_wmma_f32_16x16x32_f16(false, a, false, bc,
                                                   (short)0, acc, false, false);
    }
    // chunks 4..5: time-encoding part Phi(t) = cos(t*tw + tb)
    #pragma unroll
    for (int c = 4; c < 6; ++c) {
      const int rA = 32 * c - MEM_DIM + (hi ? 8 : 0);
      const int rB = rA + 16;
      float4 wa0 = *(const float4*)(tw + rA);
      float4 wa1 = *(const float4*)(tw + rA + 4);
      float4 wb0 = *(const float4*)(tw + rB);
      float4 wb1 = *(const float4*)(tw + rB + 4);
      float4 ba0 = *(const float4*)(tb + rA);
      float4 ba1 = *(const float4*)(tb + rA + 4);
      float4 bb0 = *(const float4*)(tb + rB);
      float4 bb1 = *(const float4*)(tb + rB + 4);
      v16h a;
      a[0]=(_Float16)__cosf(fmaf(cur_t, wa0.x, ba0.x));
      a[1]=(_Float16)__cosf(fmaf(cur_t, wa0.y, ba0.y));
      a[2]=(_Float16)__cosf(fmaf(cur_t, wa0.z, ba0.z));
      a[3]=(_Float16)__cosf(fmaf(cur_t, wa0.w, ba0.w));
      a[4]=(_Float16)__cosf(fmaf(cur_t, wa1.x, ba1.x));
      a[5]=(_Float16)__cosf(fmaf(cur_t, wa1.y, ba1.y));
      a[6]=(_Float16)__cosf(fmaf(cur_t, wa1.z, ba1.z));
      a[7]=(_Float16)__cosf(fmaf(cur_t, wa1.w, ba1.w));
      a[8]=(_Float16)__cosf(fmaf(cur_t, wb0.x, bb0.x));
      a[9]=(_Float16)__cosf(fmaf(cur_t, wb0.y, bb0.y));
      a[10]=(_Float16)__cosf(fmaf(cur_t, wb0.z, bb0.z));
      a[11]=(_Float16)__cosf(fmaf(cur_t, wb0.w, bb0.w));
      a[12]=(_Float16)__cosf(fmaf(cur_t, wb1.x, bb1.x));
      a[13]=(_Float16)__cosf(fmaf(cur_t, wb1.y, bb1.y));
      a[14]=(_Float16)__cosf(fmaf(cur_t, wb1.z, bb1.z));
      a[15]=(_Float16)__cosf(fmaf(cur_t, wb1.w, bb1.w));
      v16h bc = bwrow[2 * c];
      acc = __builtin_amdgcn_wmma_f32_16x16x32_f16(false, a, false, bc,
                                                   (short)0, acc, false, false);
    }

    // D layout: VGPR r, lanes 0-15 -> M=r; lanes 16-31 -> M=r+8 (any column).
    // Low lanes hold dots of walks 0..7, high lanes walks 8..15; one
    // shfl_xor(16) routes walk w's dot to both of its lanes.
    float mine = acc[0];
    #pragma unroll
    for (int r = 1; r < 8; ++r) if ((w & 7) == r) mine = acc[r];
    float other = __shfl_xor(mine, 16, 32);
    float dot = ((hi) == (w >= 8)) ? mine : other;

    float rho = 1.0f / (1.0f + __expf(-(dot + restart_b)));
    uint32_t hr = pcg(pcg(pcg((seed ^ 0xA5A5A5A5u) + (uint32_t)g) + (uint32_t)step));
    bool restart = u01(hr) < rho;   // identical on both lanes of the pair

    // ---- temporal-biased neighbor sampling: 16 neighbors per lane ----
    const size_t base = (size_t)cur_n * MAX_DEG + (hi ? 16 : 0);
    const float4* ntp = (const float4*)(nb_ts + base);
    const int4*   nip = (const int4*)(nb_ids + base);
    float bk = NEGF; int bid = 0; float btm = 0.0f; int bidx = 999; bool anyv = false;
    #pragma unroll
    for (int q = 0; q < 4; ++q) {
      float4 nt4 = ntp[q];
      int4   ni4 = nip[q];
      float ntv[4] = {nt4.x, nt4.y, nt4.z, nt4.w};
      int   niv[4] = {ni4.x, ni4.y, ni4.z, ni4.w};
      #pragma unroll
      for (int j = 0; j < 4; ++j) {
        int  gidx  = (hi ? 16 : 0) + q * 4 + j;
        bool valid = ntv[j] < cur_t;
        anyv |= valid;
        uint32_t h = pcg(pcg(pcg((seed ^ 0x3C6EF372u) + (uint32_t)g) +
                             (uint32_t)(step * 37)) + (uint32_t)gidx);
        float key = valid ? fmaf(ntv[j], INV_TAU, gumbel(h)) : NEGF;
        if (key > bk || (key == bk && gidx < bidx)) {
          bk = key; bid = niv[j]; btm = ntv[j]; bidx = gidx;
        }
      }
    }
    // combine the lane pair (symmetric -> both lanes agree)
    {
      float ok   = __shfl_xor(bk,   16, 32);
      int   oid  = __shfl_xor(bid,  16, 32);
      float otm  = __shfl_xor(btm,  16, 32);
      int   oidx = __shfl_xor(bidx, 16, 32);
      int   oany = __shfl_xor((int)anyv, 16, 32);
      if (ok > bk || (ok == bk && oidx < bidx)) { bk = ok; bid = oid; btm = otm; bidx = oidx; }
      anyv = anyv || (oany != 0);
    }

    int   nn  = restart ? n0  : bid;
    float ntm = restart ? t0v : btm;
    alive = alive && (restart || anyv);
    if (alive) { cur_n = nn; cur_t = ntm; }

    if (!hi) {
      size_t o = (size_t)g * L_T + step;
      float m = alive ? 1.0f : 0.0f;
      out_nodes[o] = alive ? (float)cur_n : 0.0f;
      out_times[o] = alive ? cur_t        : 0.0f;
      out_masks[o] = m;
      out_rflag[o] = (restart && alive) ? 1.0f : 0.0f;
      out_rho[o]   = rho * m;
    }
  }
}

// =================================================================
extern "C" void kernel_launch(void* const* d_in, const int* in_sizes, int n_in,
                              void* d_out, int out_size, void* d_ws, size_t ws_size,
                              hipStream_t stream) {
  const int*   src    = (const int*)  d_in[0];
  const float* t0     = (const float*)d_in[1];
  const int*   nb_ids = (const int*)  d_in[2];
  const float* nb_ts  = (const float*)d_in[3];
  const float* mem    = (const float*)d_in[4];
  const float* rw     = (const float*)d_in[5];
  const float* rb     = (const float*)d_in[6];
  const float* tw     = (const float*)d_in[7];
  const float* tb     = (const float*)d_in[8];

  float* out = (float*)d_out;

  // Output tuple layout (flat, return order), promoted to f32:
  const size_t SHORT_SZ = (size_t)BATCH * W_S * L_S;  // 245760
  const size_t LONG_SZ  = (size_t)BATCH * W_L * L_L;  // 409600
  const size_t TAWR_SZ  = (size_t)BATCH * W_T * L_T;  // 327680

  float* s_nodes = out;
  float* s_times = s_nodes + SHORT_SZ;
  float* s_masks = s_times + SHORT_SZ;
  float* l_nodes = s_masks + SHORT_SZ;
  float* l_times = l_nodes + LONG_SZ;
  float* l_masks = l_times + LONG_SZ;
  float* t_nodes = l_masks + LONG_SZ;
  float* t_times = t_nodes + TAWR_SZ;
  float* t_masks = t_times + TAWR_SZ;
  float* t_rflag = t_masks + TAWR_SZ;
  float* t_rho   = t_rflag + TAWR_SZ;

  // Short walks: 81920 walks, 1 wave each -> 10240 blocks x 256
  {
    int walks = BATCH * W_S;
    dim3 grid((walks * 32 + 255) / 256), block(256);
    plain_walks_kernel<<<grid, block, 0, stream>>>(
        src, t0, nb_ids, nb_ts, s_nodes, s_times, s_masks,
        W_S, L_S, walks, 0x13572468u);
  }
  // Long walks: 40960 walks -> 5120 blocks x 256
  {
    int walks = BATCH * W_L;
    dim3 grid((walks * 32 + 255) / 256), block(256);
    plain_walks_kernel<<<grid, block, 0, stream>>>(
        src, t0, nb_ids, nb_ts, l_nodes, l_times, l_masks,
        W_L, L_L, walks, 0x2468ACE1u);
  }
  // TAWR walks: 40960 walks, 16 per wave -> exactly 2560 full waves
  // (320 blocks x 256 threads, no partial wave -> EXEC all-ones for WMMA)
  {
    dim3 grid((BATCH * W_T / 16 * 32) / 256), block(256);
    tawr_walks_kernel<<<grid, block, 0, stream>>>(
        src, t0, nb_ids, nb_ts, mem, rw, rb, tw, tb,
        t_nodes, t_times, t_masks, t_rflag, t_rho, 0x9E3779B9u);
  }
}
